// PointerNetwork_10574209483562
// MI455X (gfx1250) — matrix operand order repaired and checked by reading, not compiled, not run on hardware
//
#include <hip/hip_runtime.h>
#include <hip/hip_bf16.h>

// ---------------------------------------------------------------------------
// Pointer network decode for MI455X (gfx1250, wave32, WMMA).
//  - enc = tokens @ We^T + be   : bf16 WMMA GEMM, result kept bf16 (L2-resident)
//  - per step: LSTM gates (i,g,o only; f is dead) via bf16 WMMA,
//              q = hx @ Wd^T + bd via bf16 WMMA,
//              scores = sum_e tanh(q+enc)*wr  via v_tanh_f32, with enc rows
//              staged LDS-side by GLOBAL_LOAD_ASYNC_TO_LDS_B128 (ASYNCcnt),
//              then log_softmax + argmax + gather next input.
// ---------------------------------------------------------------------------

typedef __bf16 bf16_t;
typedef __attribute__((ext_vector_type(16))) __bf16 bf16x16;
typedef __attribute__((ext_vector_type(8)))  float  f32x8;

#define EDIM   512
#define NTOK   512
#define BATCH  64
#define TWO_E  1024
#define FOUR_E 2048
#define STEPS  64

// ---- CDNA5 transcendental tanh (TRANS32 op) -------------------------------
__device__ inline float fast_tanh(float x) {
  float y;
  asm("v_tanh_f32 %0, %1" : "=v"(y) : "v"(x));
  return y;
}
__device__ inline float fast_sigmoid(float x) { return 0.5f * fast_tanh(0.5f * x) + 0.5f; }

// ---- CDNA5 async copy: 16B/lane, global -> LDS, tracked by ASYNCcnt -------
__device__ inline void async_g2l_b128(unsigned lds_byte, unsigned long long gaddr) {
  asm volatile("global_load_async_to_lds_b128 %0, %1, off"
               :: "v"(lds_byte), "v"(gaddr) : "memory");
}

// ---- WMMA fragment loaders (ISA 7.12.2 16-bit layouts, wave32) ------------
__device__ inline bf16x16 a_frag_f32(const float* __restrict__ A, int ld,
                                     int row, int kBase, int half) {
  const float* p = A + (size_t)row * ld + kBase + half * 8;
  bf16x16 f;
#pragma unroll
  for (int j = 0; j < 8; ++j) {
    f[j]     = (bf16_t)p[j];
    f[j + 8] = (bf16_t)p[j + 16];
  }
  return f;
}
__device__ inline bf16x16 a_frag_bf16(const bf16_t* __restrict__ A, int ld,
                                      int row, int kBase, int half) {
  const bf16_t* p = A + (size_t)row * ld + kBase + half * 8;
  bf16x16 f;
#pragma unroll
  for (int j = 0; j < 8; ++j) {
    f[j]     = p[j];
    f[j + 8] = p[j + 16];
  }
  return f;
}
__device__ inline bf16x16 b_frag(const bf16_t* __restrict__ W, int ld,
                                 int nrow, int kBase, int half) {
  const bf16_t* p = W + (size_t)nrow * ld + kBase + half * 8;
  bf16x16 f;
#pragma unroll
  for (int j = 0; j < 8; ++j) {
    f[j]     = p[j];
    f[j + 8] = p[j + 16];
  }
  return f;
}
__device__ inline f32x8 wmma_bf16(bf16x16 a, bf16x16 b, f32x8 c) {
  return __builtin_amdgcn_wmma_f32_16x16x32_bf16(false, a, false, b, (short)0, c,
                                                 false, false);
}

// ---- small helpers --------------------------------------------------------
__global__ void cvt_kernel(const float* __restrict__ src, bf16_t* __restrict__ dst, int n) {
  int i = blockIdx.x * blockDim.x + threadIdx.x;
  if (i < n) dst[i] = (bf16_t)src[i];
}
__global__ void copy_kernel(const float* __restrict__ src, float* __restrict__ dst, int n) {
  int i = blockIdx.x * blockDim.x + threadIdx.x;
  if (i < n) dst[i] = src[i];
}

// ---- encoder GEMM: enc[m,e] = sum_k tokens[m,k]*We[e,k] + be[e] -----------
__global__ __launch_bounds__(256) void encoder_kernel(
    const float* __restrict__ tokens, const bf16_t* __restrict__ WeB,
    const float* __restrict__ be, bf16_t* __restrict__ enc) {
  __shared__ bf16_t As[16 * TWO_E];  // 32 KB
  const int tid = threadIdx.x;
  const int mTile = blockIdx.x;
  const float* src = tokens + (size_t)mTile * 16 * TWO_E;
#pragma unroll
  for (int i = 0; i < 64; ++i) {
    int idx = tid + i * 256;
    As[idx] = (bf16_t)src[idx];
  }
  __syncthreads();

  const int wave = tid >> 5, lane = tid & 31, l16 = lane & 15, half = lane >> 4;
  f32x8 acc[4];
#pragma unroll
  for (int j = 0; j < 4; ++j)
#pragma unroll
    for (int r = 0; r < 8; ++r) acc[j][r] = 0.f;

  for (int kt = 0; kt < 32; ++kt) {
    const int kBase = kt * 32;
    bf16x16 a = a_frag_bf16(As, TWO_E, l16, kBase, half);
#pragma unroll
    for (int j = 0; j < 4; ++j) {
      const int nTile = wave + 8 * j;
      bf16x16 b = b_frag(WeB, TWO_E, nTile * 16 + l16, kBase, half);
      acc[j] = wmma_bf16(a, b, acc[j]);
    }
  }
#pragma unroll
  for (int j = 0; j < 4; ++j) {
    const int eBase = (wave + 8 * j) * 16;
    const float bias = be[eBase + l16];
#pragma unroll
    for (int r = 0; r < 8; ++r) {
      const int m = mTile * 16 + r + half * 8;
      enc[(size_t)m * EDIM + eBase + l16] = (bf16_t)(acc[j][r] + bias);
    }
  }
}

// ---- LSTM gates (zero hidden state; f gate dead) --------------------------
__global__ __launch_bounds__(256) void lstm_kernel(
    const float* __restrict__ dec_in, const bf16_t* __restrict__ WihB,
    const float* __restrict__ b_ih, const float* __restrict__ b_hh,
    bf16_t* __restrict__ hx) {
  const int tid = threadIdx.x;
  const int wave = tid >> 5, lane = tid & 31, l16 = lane & 15, half = lane >> 4;
  const int g = blockIdx.x * 8 + wave;
  const int mTile = g & 3, eTile = g >> 2;

  f32x8 ai, ag, ao;
#pragma unroll
  for (int r = 0; r < 8; ++r) { ai[r] = 0.f; ag[r] = 0.f; ao[r] = 0.f; }

  const int rowI = 0 * EDIM + eTile * 16 + l16;
  const int rowG = 2 * EDIM + eTile * 16 + l16;
  const int rowO = 3 * EDIM + eTile * 16 + l16;
  const int aRow = mTile * 16 + l16;

  for (int kt = 0; kt < 32; ++kt) {
    const int kBase = kt * 32;
    bf16x16 a = a_frag_f32(dec_in, TWO_E, aRow, kBase, half);
    ai = wmma_bf16(a, b_frag(WihB, TWO_E, rowI, kBase, half), ai);
    ag = wmma_bf16(a, b_frag(WihB, TWO_E, rowG, kBase, half), ag);
    ao = wmma_bf16(a, b_frag(WihB, TWO_E, rowO, kBase, half), ao);
  }
  const int e = eTile * 16 + l16;
  const float bi = b_ih[e] + b_hh[e];
  const float bg = b_ih[2 * EDIM + e] + b_hh[2 * EDIM + e];
  const float bo = b_ih[3 * EDIM + e] + b_hh[3 * EDIM + e];
#pragma unroll
  for (int r = 0; r < 8; ++r) {
    const int m = mTile * 16 + r + half * 8;
    const float c = fast_sigmoid(ai[r] + bi) * fast_tanh(ag[r] + bg);
    const float h = fast_sigmoid(ao[r] + bo) * fast_tanh(c);
    hx[(size_t)m * EDIM + e] = (bf16_t)h;
  }
}

// ---- q = hx @ Wd^T + bd ---------------------------------------------------
__global__ __launch_bounds__(256) void qproj_kernel(
    const bf16_t* __restrict__ hx, const bf16_t* __restrict__ WdB,
    const float* __restrict__ bd, float* __restrict__ q) {
  const int tid = threadIdx.x;
  const int wave = tid >> 5, lane = tid & 31, l16 = lane & 15, half = lane >> 4;
  const int g = blockIdx.x * 8 + wave;
  const int mTile = g & 3, eTile = g >> 2;

  f32x8 acc;
#pragma unroll
  for (int r = 0; r < 8; ++r) acc[r] = 0.f;

  for (int kt = 0; kt < 16; ++kt) {
    const int kBase = kt * 32;
    bf16x16 a = a_frag_bf16(hx, EDIM, mTile * 16 + l16, kBase, half);
    bf16x16 b = b_frag(WdB, EDIM, eTile * 16 + l16, kBase, half);
    acc = wmma_bf16(a, b, acc);
  }
  const int e = eTile * 16 + l16;
  const float bias = bd[e];
#pragma unroll
  for (int r = 0; r < 8; ++r) {
    const int m = mTile * 16 + r + half * 8;
    q[(size_t)m * EDIM + e] = acc[r] + bias;
  }
}

// ---- scores[b,n] = sum_e tanh(q[b,e]+enc[b,n,e])*wr[e] + br ---------------
// grid = (4 chunks, 64 batch) x 256 thr (8 waves). Wave handles 16 rows,
// double-buffered through LDS via async global->LDS copies (ASYNCcnt).
__global__ __launch_bounds__(256) void scores_kernel(
    const bf16_t* __restrict__ enc, const float* __restrict__ q,
    const float* __restrict__ wr, const float* __restrict__ br,
    float* __restrict__ scores) {
  __shared__ float q_s[EDIM];
  __shared__ float wr_s[EDIM];
  __shared__ __align__(16) bf16_t ebuf[8 * 2 * EDIM];  // 16 KB: 2 row-buffers/wave

  const int b = blockIdx.y, chunk = blockIdx.x, tid = threadIdx.x;
  const int wave = tid >> 5, lane = tid & 31;

  q_s[tid]        = q[(size_t)b * EDIM + tid];
  q_s[tid + 256]  = q[(size_t)b * EDIM + tid + 256];
  wr_s[tid]       = wr[tid];
  wr_s[tid + 256] = wr[tid + 256];
  __syncthreads();

  const int nBase = chunk * 128 + wave * 16;
  const unsigned long long rowBase =
      (unsigned long long)(uintptr_t)(enc + ((size_t)(b * NTOK + nBase)) * EDIM);
  const unsigned bufByte0 =
      (unsigned)(uintptr_t)(&ebuf[(wave * 2 + 0) * EDIM]) + (unsigned)lane * 16u;
  const unsigned bufByte1 =
      (unsigned)(uintptr_t)(&ebuf[(wave * 2 + 1) * EDIM]) + (unsigned)lane * 16u;

  // prologue: kick off row 0 (two 512B async transfers)
  {
    unsigned long long g0 = rowBase + (unsigned long long)lane * 16ull;
    async_g2l_b128(bufByte0, g0);
    async_g2l_b128(bufByte0 + 512u, g0 + 512ull);
  }
  const float br0 = br[0];

  for (int i = 0; i < 16; ++i) {
    if (i < 15) {  // issue next row into the other buffer, keep 2 in flight
      unsigned long long g =
          rowBase + (unsigned long long)(i + 1) * (EDIM * 2ull) +
          (unsigned long long)lane * 16ull;
      unsigned dst = ((i + 1) & 1) ? bufByte1 : bufByte0;
      async_g2l_b128(dst, g);
      async_g2l_b128(dst + 512u, g + 512ull);
      asm volatile("s_wait_asynccnt 0x2" ::: "memory");  // row i landed
    } else {
      asm volatile("s_wait_asynccnt 0x0" ::: "memory");
    }
    const bf16_t* row = &ebuf[(wave * 2 + (i & 1)) * EDIM];
    float s = 0.f;
#pragma unroll
    for (int it = 0; it < 4; ++it) {
      const int e0 = it * 128 + lane * 4;
      uint2 w = *(const uint2*)(row + e0);
      const float v0 = __uint_as_float((w.x & 0xffffu) << 16);
      const float v1 = __uint_as_float(w.x & 0xffff0000u);
      const float v2 = __uint_as_float((w.y & 0xffffu) << 16);
      const float v3 = __uint_as_float(w.y & 0xffff0000u);
      s += fast_tanh(q_s[e0 + 0] + v0) * wr_s[e0 + 0];
      s += fast_tanh(q_s[e0 + 1] + v1) * wr_s[e0 + 1];
      s += fast_tanh(q_s[e0 + 2] + v2) * wr_s[e0 + 2];
      s += fast_tanh(q_s[e0 + 3] + v3) * wr_s[e0 + 3];
    }
#pragma unroll
    for (int m = 16; m > 0; m >>= 1) s += __shfl_xor(s, m, 32);
    if (lane == 0) scores[(size_t)b * NTOK + nBase + i] = s + br0;
  }
}

// ---- log_softmax + argmax + gather (one block per batch row) --------------
__global__ __launch_bounds__(512) void finalize_kernel(
    const float* __restrict__ scores, const float* __restrict__ tokens,
    float* __restrict__ dec_in, float* __restrict__ out_probs,
    float* __restrict__ out_idx, int step) {
  __shared__ float sc_s[NTOK];
  __shared__ float rmax_s[16];
  __shared__ int   ridx_s[16];
  __shared__ float rsum_s[16];
  __shared__ float maxv_s, logZ_s;
  __shared__ int   best_s;

  const int b = blockIdx.x, tid = threadIdx.x;
  const int wave = tid >> 5, lane = tid & 31;
  sc_s[tid] = scores[(size_t)b * NTOK + tid];
  __syncthreads();

  // argmax (first-max tie-break, like jnp.argmax)
  float v = sc_s[tid];
  int   idx = tid;
#pragma unroll
  for (int m = 16; m > 0; m >>= 1) {
    float ov = __shfl_xor(v, m, 32);
    int   oi = __shfl_xor(idx, m, 32);
    if (ov > v || (ov == v && oi < idx)) { v = ov; idx = oi; }
  }
  if (lane == 0) { rmax_s[wave] = v; ridx_s[wave] = idx; }
  __syncthreads();
  if (tid < 32) {
    float v2 = (tid < 16) ? rmax_s[tid] : -3.4e38f;
    int   i2 = (tid < 16) ? ridx_s[tid] : 0x7fffffff;
#pragma unroll
    for (int m = 8; m > 0; m >>= 1) {
      float ov = __shfl_xor(v2, m, 32);
      int   oi = __shfl_xor(i2, m, 32);
      if (ov > v2 || (ov == v2 && oi < i2)) { v2 = ov; i2 = oi; }
    }
    if (tid == 0) { maxv_s = v2; best_s = i2; }
  }
  __syncthreads();

  const float maxv = maxv_s;
  float ssum = __expf(sc_s[tid] - maxv);
#pragma unroll
  for (int m = 16; m > 0; m >>= 1) ssum += __shfl_xor(ssum, m, 32);
  if (lane == 0) rsum_s[wave] = ssum;
  __syncthreads();
  if (tid < 16) {
    float t = rsum_s[tid];
#pragma unroll
    for (int m = 8; m > 0; m >>= 1) t += __shfl_xor(t, m, 32);
    if (tid == 0) logZ_s = __logf(t);
  }
  __syncthreads();

  out_probs[((size_t)(b * STEPS + step)) * NTOK + tid] = sc_s[tid] - maxv - logZ_s;
  const int best = best_s;
  if (tid == 0) out_idx[b * STEPS + step] = (float)best;

  const float* trow = tokens + ((size_t)(b * NTOK + best)) * TWO_E;
  dec_in[(size_t)b * TWO_E + tid]       = trow[tid];
  dec_in[(size_t)b * TWO_E + tid + 512] = trow[tid + 512];
}

// ---------------------------------------------------------------------------
extern "C" void kernel_launch(void* const* d_in, const int* in_sizes, int n_in,
                              void* d_out, int out_size, void* d_ws, size_t ws_size,
                              hipStream_t stream) {
  (void)in_sizes; (void)n_in; (void)out_size; (void)ws_size;
  const float* tokens  = (const float*)d_in[0];
  const float* enc_doc = (const float*)d_in[1];
  const float* W_ih    = (const float*)d_in[2];
  const float* b_ih    = (const float*)d_in[3];
  const float* b_hh    = (const float*)d_in[4];
  const float* Wd      = (const float*)d_in[5];
  const float* bd      = (const float*)d_in[6];
  const float* We      = (const float*)d_in[7];
  const float* be      = (const float*)d_in[8];
  const float* wr      = (const float*)d_in[9];
  const float* br      = (const float*)d_in[10];

  float* out_probs = (float*)d_out;
  float* out_idx   = out_probs + (size_t)BATCH * STEPS * NTOK;

  char* ws = (char*)d_ws;
  size_t off = 0;
  auto walloc = [&](size_t bytes) -> void* {
    void* p = ws + off;
    off += (bytes + 255) & ~(size_t)255;
    return p;
  };
  bf16_t* encB = (bf16_t*)walloc((size_t)BATCH * NTOK * EDIM * 2);  // 32 MB, L2-resident
  bf16_t* WihB = (bf16_t*)walloc((size_t)FOUR_E * TWO_E * 2);       // 4 MB
  bf16_t* WdB  = (bf16_t*)walloc((size_t)EDIM * EDIM * 2);
  bf16_t* WeB  = (bf16_t*)walloc((size_t)EDIM * TWO_E * 2);
  bf16_t* hxB  = (bf16_t*)walloc((size_t)BATCH * EDIM * 2);
  float*  dec  = (float*)walloc((size_t)BATCH * TWO_E * 4);
  float*  qbuf = (float*)walloc((size_t)BATCH * EDIM * 4);
  float*  scW  = (float*)walloc((size_t)BATCH * NTOK * 4);

  cvt_kernel<<<(FOUR_E * TWO_E + 255) / 256, 256, 0, stream>>>(W_ih, WihB, FOUR_E * TWO_E);
  cvt_kernel<<<(EDIM * EDIM + 255) / 256, 256, 0, stream>>>(Wd, WdB, EDIM * EDIM);
  cvt_kernel<<<(EDIM * TWO_E + 255) / 256, 256, 0, stream>>>(We, WeB, EDIM * TWO_E);
  copy_kernel<<<(BATCH * TWO_E + 255) / 256, 256, 0, stream>>>(enc_doc, dec, BATCH * TWO_E);

  encoder_kernel<<<BATCH * NTOK / 16, 256, 0, stream>>>(tokens, WeB, be, encB);

  for (int t = 0; t < STEPS; ++t) {
    lstm_kernel<<<16, 256, 0, stream>>>(dec, WihB, b_ih, b_hh, hxB);
    qproj_kernel<<<16, 256, 0, stream>>>(hxB, WdB, bd, qbuf);
    scores_kernel<<<dim3(4, BATCH), 256, 0, stream>>>(encB, qbuf, wr, br, scW);
    finalize_kernel<<<BATCH, 512, 0, stream>>>(scW, tokens, dec, out_probs, out_idx, t);
  }
}
